// BoundaryAwareLoss_79027398246745
// MI455X (gfx1250) — compile-verified, matrix operand order checked by reading, not run on hardware
//
#include <hip/hip_runtime.h>

#define BATCH 2
#define NCLS  19
#define DEMB  32
#define KSEG  16
#define HH    512
#define WW    1024
#define HWPIX (HH * WW)   // 524288 pixels per batch
#define IGN   255

typedef __attribute__((ext_vector_type(16))) _Float16 v16h;
typedef __attribute__((ext_vector_type(8)))  float    v8f;
typedef __attribute__((ext_vector_type(8)))  short    v8s;
typedef __attribute__((ext_vector_type(4)))  unsigned int v4u;
typedef __attribute__((ext_vector_type(8)))  int      v8i;
typedef __attribute__((ext_vector_type(4)))  int      v4i;

#if __has_builtin(__builtin_amdgcn_tensor_load_to_lds)
#define USE_TDM 1
#else
#define USE_TDM 0
#endif

// -------- workspace layout (floats) --------
#define WS_CNT     0      // [16]  per-segment counts
#define WS_SUM     16     // [16*32] per-segment embedding sums
#define WS_CEN     528    // [16*32] centers
#define WS_VAR     1040   // [16]  hinge sums per segment
#define WS_DNUM    1056   // push-term numerator
#define WS_NPAIR   1057   // number of pairs
#define WS_HAS     1058   // has_pairs flag
#define WS_BSTRIDE 1088
#define WS_SEMN    (2 * WS_BSTRIDE + 0)  // sum(nll * weight)
#define WS_PW      (2 * WS_BSTRIDE + 1)  // sum(weight)
#define WS_TOTAL_FLOATS (2 * WS_BSTRIDE + 8)

// =====================================================================
// Kernel 1: Sobel boundary + weighted CE, block-reduced partial sums
// =====================================================================
__global__ void k_semantic(const float* __restrict__ logits,
                           const int* __restrict__ labels,
                           float* __restrict__ ws) {
  __shared__ float r0[256], r1[256];
  const int tid = threadIdx.x;
  float wnll = 0.f, wsum = 0.f;
  const long total = (long)BATCH * HWPIX;
  for (long p = (long)blockIdx.x * blockDim.x + tid; p < total;
       p += (long)gridDim.x * blockDim.x) {
    const int b  = (int)(p / HWPIX);
    const int hw = (int)(p - (long)b * HWPIX);
    const int y = hw / WW, x = hw - y * WW;
    const int* lb = labels + (long)b * HWPIX;
    // 3x3 zero-padded label neighborhood; fast path for interior pixels
    float v[3][3];
    if (y > 0 && y < HH - 1 && x > 0 && x < WW - 1) {
      const int* c = lb + hw;
#pragma unroll
      for (int dy = -1; dy <= 1; ++dy)
#pragma unroll
        for (int dx = -1; dx <= 1; ++dx)
          v[dy + 1][dx + 1] = (float)c[dy * WW + dx];
    } else {
#pragma unroll
      for (int dy = -1; dy <= 1; ++dy)
#pragma unroll
        for (int dx = -1; dx <= 1; ++dx) {
          const int yy = y + dy, xx = x + dx;
          v[dy + 1][dx + 1] = (yy >= 0 && yy < HH && xx >= 0 && xx < WW)
                                  ? (float)lb[yy * WW + xx] : 0.f;
        }
    }
    const float gx = -v[0][0] + v[0][2] - 2.f * v[1][0] + 2.f * v[1][2]
                     - v[2][0] + v[2][2];
    const float gy = -v[0][0] - 2.f * v[0][1] - v[0][2]
                     + v[2][0] + 2.f * v[2][1] + v[2][2];
    const float boundary = (gx * gx + gy * gy > 0.01f) ? 1.f : 0.f;

    const int lab = lb[hw];
    const float valid = (lab != IGN) ? 1.f : 0.f;
    const int slab = (lab == IGN) ? 0 : lab;
    const float w = valid * (1.f + boundary);   // BOUNDARY_WEIGHT = 2

    // log-sum-exp over 19 classes: load all, max-reduce, exp-sum (good ILP)
    const float* lg = logits + ((long)b * NCLS) * HWPIX + hw;
    float xv[NCLS];
#pragma unroll
    for (int c = 0; c < NCLS; ++c) xv[c] = lg[(long)c * HWPIX];
    float m = xv[0], xl = xv[0];
#pragma unroll
    for (int c = 1; c < NCLS; ++c) {
      m = fmaxf(m, xv[c]);
      xl = (c == slab) ? xv[c] : xl;   // cndmask, avoids dynamic reg index
    }
    float s = 0.f;
#pragma unroll
    for (int c = 0; c < NCLS; ++c) s += __expf(xv[c] - m);
    const float nll = (m + __logf(s)) - xl;
    wnll += nll * w;
    wsum += w;
  }
  r0[tid] = wnll; r1[tid] = wsum;
  __syncthreads();
  for (int off = 128; off > 0; off >>= 1) {
    if (tid < off) { r0[tid] += r0[tid + off]; r1[tid] += r1[tid + off]; }
    __syncthreads();
  }
  if (tid == 0) {
    atomicAdd(&ws[WS_SEMN], r0[0]);
    atomicAdd(&ws[WS_PW],   r1[0]);
  }
}

// =====================================================================
// Kernel 2: segment counts + sums via V_WMMA_F32_16X16X32_F16
//   sums[16,32] = onehot(labels)[16,N] x emb[N,32], 32 pixels per WMMA pair.
//   Double-buffered TDM pipeline: wave 0 issues tensor_load_to_lds for
//   tile t+1 into the spare buffer, then s_wait_tensorcnt(1) so only the
//   current tile's DMA is waited on — the next transfer streams in behind
//   the ds_load_b128 gathers + WMMAs.
// =====================================================================
#define TILE  256   // pixels per block iteration (8 waves * 32)
#define PITCH 260   // floats per dim row in LDS (256 + 4 pad dwords)

#if USE_TDM
__device__ __forceinline__ void tdm_issue_tile(const float* gsrc,
                                               const float* lds_dst) {
  // 2D Tensor DMA descriptor: 32 rows x 256 f32, LDS rows padded to 260
  const unsigned long long ga = (unsigned long long)(const void*)gsrc;
  v4u g0;
  g0[0] = 1u;                                   // count=1, load, user D#
  g0[1] = (unsigned)(unsigned long long)(const void*)lds_dst; // lds_addr
  g0[2] = (unsigned)(ga & 0xFFFFFFFFull);       // global_addr[31:0]
  g0[3] = (unsigned)((ga >> 32) & 0x1FFFFFFull) // global_addr[56:32]
          | (2u << 30);                         // type = 2 (image)
  v8i g1;
  g1[0] = (int)((2u << 16)        // data_size = 4B
                | (1u << 20)      // pad_enable
                | (7u << 22)      // pad_interval: every 256 dwords
                | (3u << 25));    // pad_amount: 4 dwords -> pitch 260
  g1[1] = 0;                                    // no atomic barrier
  g1[2] = (int)(((unsigned)HWPIX >> 16)         // tensor_dim0[31:16]
                | (32u << 16));                 // tensor_dim1[15:0]
  g1[3] = (int)(256u << 16);                    // tile_dim0 = 256
  g1[4] = 32;                                   // tile_dim1 = 32
  g1[5] = (int)(unsigned)HWPIX;                 // tensor_dim0_stride lo32
  g1[6] = 0;
  g1[7] = 0;
  v4i gz = {0, 0, 0, 0};                        // groups 2/3 unused (2D)
#if __clang_major__ >= 23
  v8i gz8 = {};
  __builtin_amdgcn_tensor_load_to_lds(g0, g1, gz, gz, gz8, 0);
#else
  __builtin_amdgcn_tensor_load_to_lds(g0, g1, gz, gz, 0);
#endif
}
#endif

__global__ void k_segsum_wmma(const float* __restrict__ emb,
                              const int* __restrict__ labels,
                              float* __restrict__ ws) {
  __shared__ __align__(16) float sh_f[2][DEMB * PITCH];  // 2 x 33280 B
  __shared__ __align__(16) short sh_ls[TILE];            // labels as i16
  __shared__ float sh_cnt[KSEG];

  const int b = blockIdx.y;
  const float* e  = emb + (long)b * DEMB * HWPIX;
  const int*   lb = labels + (long)b * HWPIX;
  float* g_cnt = ws + (long)b * WS_BSTRIDE + WS_CNT;
  float* g_sum = ws + (long)b * WS_BSTRIDE + WS_SUM;

  const int tid  = threadIdx.x;
  const int lane = tid & 31;
  const int wv   = tid >> 5;
  const int row  = lane & 15;              // M for A-matrix, N for B-matrix
  const int hi8  = (lane < 16) ? 0 : 8;    // K-group offset per lane half
  if (tid < KSEG) sh_cnt[tid] = 0.f;

  v8f acc0 = {}, acc1 = {};
  const int stride = gridDim.x * TILE;     // HWPIX divisible by stride
  int cur = 0;
#if USE_TDM
  if (wv == 0) tdm_issue_tile(e + blockIdx.x * TILE, sh_f[0]);  // prologue
#endif
  for (int tile = blockIdx.x * TILE; tile < HWPIX; tile += stride) {
    // labels: stage as i16 + count histogram (tile+tid < HWPIX by grid sizing)
    const int p = tile + tid;
    const int lv = lb[p];
    sh_ls[tid] = (short)lv;
    if ((unsigned)lv < KSEG) atomicAdd(&sh_cnt[lv], 1.f);
    __builtin_prefetch(&lb[p + stride], 0, 1);      // global_prefetch_b8
#if USE_TDM
    if (wv == 0) {
      const int nxt = tile + stride;
      if (nxt < HWPIX) {
        tdm_issue_tile(e + nxt, sh_f[cur ^ 1]);     // overlap next DMA
        __builtin_amdgcn_s_wait_tensorcnt(1);       // wait current only
      } else {
        __builtin_amdgcn_s_wait_tensorcnt(0);
      }
    }
#else
    for (int i = tid; i < DEMB * TILE; i += 256) {
      const int d = i >> 8, pp = i & 255;
      sh_f[cur][d * PITCH + pp] = e[(long)d * HWPIX + tile + pp];
    }
#endif
    __syncthreads();   // tile data + labels visible to all waves

    // ---- gather WMMA operands: contiguous 8-pixel runs (ds_load_b128) ----
    const int r0 = wv * 32 + hi8;                   // first run of 8 pixels
    const float* fr0 = &sh_f[cur][row * PITCH + r0];
    const float* fr1 = &sh_f[cur][(row + 16) * PITCH + r0];
    const v8f f00 = *(const v8f*)(fr0);             // dims 0-15 operand halves
    const v8f f01 = *(const v8f*)(fr0 + 16);
    const v8f f10 = *(const v8f*)(fr1);             // dims 16-31
    const v8f f11 = *(const v8f*)(fr1 + 16);
    const v8s s0  = *(const v8s*)&sh_ls[r0];
    const v8s s1  = *(const v8s*)&sh_ls[r0 + 16];
    v16h A, B0, B1;
#pragma unroll
    for (int h = 0; h < 8; ++h) {
      A[h]      = (s0[h] == (short)row) ? (_Float16)1.0f : (_Float16)0.0f;
      A[h + 8]  = (s1[h] == (short)row) ? (_Float16)1.0f : (_Float16)0.0f;
      B0[h]     = (_Float16)f00[h];
      B0[h + 8] = (_Float16)f01[h];
      B1[h]     = (_Float16)f10[h];
      B1[h + 8] = (_Float16)f11[h];
    }
    acc0 = __builtin_amdgcn_wmma_f32_16x16x32_f16(false, A, false, B0,
                                                  (short)0, acc0, false, false);
    acc1 = __builtin_amdgcn_wmma_f32_16x16x32_f16(false, A, false, B1,
                                                  (short)0, acc1, false, false);
    cur ^= 1;
    __syncthreads();   // consumers done before this buffer is refilled
  }
  // ---- flush: C/D layout VGPR i, lanes<16 -> M=i, lanes>=16 -> M=i+8 ----
#pragma unroll
  for (int i = 0; i < 8; ++i) {
    const int Mo = i + hi8;
    atomicAdd(&g_sum[Mo * DEMB + row],      acc0[i]);
    atomicAdd(&g_sum[Mo * DEMB + row + 16], acc1[i]);
  }
  if (tid < KSEG) atomicAdd(&g_cnt[tid], sh_cnt[tid]);
}

// =====================================================================
// Kernel 3: centers = sums / max(counts,1); push (distance) term per batch
// =====================================================================
__global__ void k_centers_push(float* __restrict__ ws) {
  const int b = blockIdx.x;
  float* base  = ws + (long)b * WS_BSTRIDE;
  float* g_cnt = base + WS_CNT;
  float* g_sum = base + WS_SUM;
  float* g_cen = base + WS_CEN;
  __shared__ float sh_c[KSEG * DEMB];
  __shared__ float sh_cnt[KSEG];
  __shared__ int   sh_pres[KSEG];
  __shared__ float sh_dh, sh_np;
  const int tid = threadIdx.x;
  if (tid == 0) { sh_dh = 0.f; sh_np = 0.f; }
  if (tid < KSEG) {
    const float c = g_cnt[tid];
    sh_cnt[tid]  = c;
    sh_pres[tid] = (c > 0.f && tid != 0) ? 1 : 0;  // exclude background 0
  }
  __syncthreads();
  for (int i = tid; i < KSEG * DEMB; i += blockDim.x) {
    const int k = i / DEMB;
    const float c = g_sum[i] / fmaxf(sh_cnt[k], 1.f);
    sh_c[i]  = c;
    g_cen[i] = c;
  }
  __syncthreads();
  {  // all 256 ordered pairs (i,j)
    const int i = tid >> 4, j = tid & 15;
    if (i != j && sh_pres[i] && sh_pres[j]) {
      float sq = 0.f;
#pragma unroll
      for (int d = 0; d < DEMB; ++d) {
        const float df = sh_c[i * DEMB + d] - sh_c[j * DEMB + d];
        sq += df * df;
      }
      const float pd = sqrtf(sq);
      const float hd = fmaxf(3.0f - pd, 0.f);   // 2*DELTA_D = 3.0
      atomicAdd(&sh_dh, hd * hd);
      atomicAdd(&sh_np, 1.f);
    }
  }
  __syncthreads();
  if (tid == 0) {
    int np = 0;
    for (int k = 0; k < KSEG; ++k) np += sh_pres[k];
    base[WS_DNUM]  = sh_dh;
    base[WS_NPAIR] = sh_np;
    base[WS_HAS]   = (np > 1) ? 1.f : 0.f;
  }
}

// =====================================================================
// Kernel 4: pull (variance) term — per-pixel hinge vs center[label]
//   (instance_logits resident in 192MB L2 after pass A)
// =====================================================================
__global__ void k_variance(const float* __restrict__ emb,
                           const int* __restrict__ labels,
                           float* __restrict__ ws) {
  const int b = blockIdx.y;
  const float* e  = emb + (long)b * DEMB * HWPIX;
  const int*   lb = labels + (long)b * HWPIX;
  float* base = ws + (long)b * WS_BSTRIDE;
  const float* g_cen = base + WS_CEN;
  float* g_var = base + WS_VAR;
  __shared__ float sh_c[KSEG * DEMB];
  __shared__ float sh_v[KSEG];
  const int tid = threadIdx.x;
  for (int i = tid; i < KSEG * DEMB; i += blockDim.x) sh_c[i] = g_cen[i];
  if (tid < KSEG) sh_v[tid] = 0.f;
  __syncthreads();
  for (int p = blockIdx.x * blockDim.x + tid; p < HWPIX;
       p += gridDim.x * blockDim.x) {
    const int lv = lb[p];
    if (lv > 0 && lv < KSEG) {   // background masked by `present` anyway
      float ss = 0.f;
#pragma unroll
      for (int d = 0; d < DEMB; ++d) {
        const float df = e[(long)d * HWPIX + p] - sh_c[lv * DEMB + d];
        ss += df * df;
      }
      const float dist = sqrtf(fmaxf(ss, 1e-12f));
      const float h = fmaxf(dist - 0.5f, 0.f);  // DELTA_V = 0.5
      atomicAdd(&sh_v[lv], h * h);
    }
  }
  __syncthreads();
  if (tid < KSEG) atomicAdd(&g_var[tid], sh_v[tid]);
}

// =====================================================================
// Kernel 5: final combine -> [total, semantic, instance, mean(pixel_weights)]
// =====================================================================
__global__ void k_final(const float* __restrict__ ws, float* __restrict__ out) {
  if (threadIdx.x != 0 || blockIdx.x != 0) return;
  const float swnll = ws[WS_SEMN], spw = ws[WS_PW];
  const float sem = swnll / (spw + 1e-8f);
  const float pw_mean = spw / (float)((long)BATCH * HWPIX);
  float vs = 0.f, nv = 0.f, dsum = 0.f, nd = 0.f;
  for (int b = 0; b < BATCH; ++b) {
    const float* base = ws + (long)b * WS_BSTRIDE;
    for (int k = 1; k < KSEG; ++k) {
      const float c = base[WS_CNT + k];
      if (c > 0.f) {
        vs += base[WS_VAR + k] / fmaxf(c, 1.f);
        nv += 1.f;
      }
    }
    if (base[WS_HAS] > 0.f) {
      dsum += base[WS_DNUM] / fmaxf(base[WS_NPAIR], 1.f);
      nd += 1.f;
    }
  }
  const float loss_var  = vs / fmaxf(nv, 1.f);
  const float loss_dist = dsum / fmaxf(nd, 1.f);
  const float inst  = loss_var + loss_dist;
  const float total = sem + inst;   // SEMANTIC_WEIGHT = INSTANCE_WEIGHT = 1
  out[0] = total; out[1] = sem; out[2] = inst; out[3] = pw_mean;
}

extern "C" void kernel_launch(void* const* d_in, const int* in_sizes, int n_in,
                              void* d_out, int out_size, void* d_ws, size_t ws_size,
                              hipStream_t stream) {
  const float* sem_logits  = (const float*)d_in[0];
  const float* inst_logits = (const float*)d_in[1];
  const int*   sem_labels  = (const int*)d_in[2];
  const int*   inst_labels = (const int*)d_in[3];
  float* out = (float*)d_out;
  float* ws  = (float*)d_ws;

  hipMemsetAsync(d_ws, 0, WS_TOTAL_FLOATS * sizeof(float), stream);

  k_semantic   <<<dim3(1024),        dim3(256), 0, stream>>>(sem_logits, sem_labels, ws);
  k_segsum_wmma<<<dim3(512, BATCH),  dim3(256), 0, stream>>>(inst_logits, inst_labels, ws);
  k_centers_push<<<dim3(BATCH),      dim3(256), 0, stream>>>(ws);
  k_variance   <<<dim3(1024, BATCH), dim3(256), 0, stream>>>(inst_logits, inst_labels, ws);
  k_final      <<<dim3(1),           dim3(32),  0, stream>>>(ws, out);
}
